// Attention_3375844295015
// MI455X (gfx1250) — matrix operand order repaired and checked by reading, not compiled, not run on hardware
//
#include <hip/hip_runtime.h>

typedef __attribute__((ext_vector_type(16))) _Float16 v16h;
typedef __attribute__((ext_vector_type(8)))  _Float16 h8;
typedef __attribute__((ext_vector_type(8)))  float    v8f;

#define BQ 2
#define NQ 2048
#define DQ 1024
#define HQ 16
#define DHQ 64
#define MROWS (BQ*NQ)   // 4096

// ---------------- CDNA5 async global->LDS copy (ASYNCcnt path) ----------------
// LDS destination address comes from a VGPR; we pass the low 32 bits of the generic LDS
// address (flat LDS addressing uses addr[31:0]).

__device__ __forceinline__ unsigned lds_addr32(const void* p) {
  return (unsigned)(size_t)p;
}

__device__ __forceinline__ void async_copy_b128(unsigned lds_off,
                                                const void* gaddr) {
  asm volatile("global_load_async_to_lds_b128 %0, %1, off"
               :: "v"(lds_off), "v"(gaddr)
               : "memory");
}

__device__ __forceinline__ void async_wait0() {
  asm volatile("s_wait_asynccnt 0x0" ::: "memory");
}

// ---------------- WMMA helpers (gfx1250 layouts, wave32) ----------------

__device__ __forceinline__ v8f wmma32(v16h a, v16h b, v8f c) {
  // D = A(16x32 f16) * B(32x16 f16) + C(16x16 f32)
  return __builtin_amdgcn_wmma_f32_16x16x32_f16(false, a, false, b, (short)0, c,
                                                false, false);
}

// A fragment 16(M)x32(K), src row-major [m][k], stride ld (halfs).
// lane<16: m=lane, K in [kb,kb+16) with kb = (lane>>4)*16
__device__ __forceinline__ v16h load_frag_a(const _Float16* src, int ld) {
  int lane = threadIdx.x & 31;
  int m = lane & 15;
  int kb = (lane >> 4) * 16;
  const _Float16* p = src + (size_t)m * ld + kb;
  v16h o;
#pragma unroll
  for (int j = 0; j < 8; ++j) o[j] = p[j];
#pragma unroll
  for (int j = 0; j < 8; ++j) o[8 + j] = p[8 + j];
  return o;
}

// B fragment 32(K)x16(N) from a transposed source: B(k,n) = src[n][k].
// Contiguous 16-half run per lane -> 2x ds_load_b128.
// lanes 0-15 hold K=0..15, lanes 16-31 hold K=16..31, N = lane&15
__device__ __forceinline__ v16h load_frag_bT(const _Float16* src, int ld) {
  int lane = threadIdx.x & 31;
  int n = lane & 15;
  int kb = (lane >> 4) * 16;
  const _Float16* p = src + (size_t)n * ld + kb;
  v16h o;
#pragma unroll
  for (int j = 0; j < 16; ++j) o[j] = p[j];
  return o;
}

// ---------------- elementwise / prep kernels ----------------

__global__ void f32_to_f16_kernel(const float* __restrict__ src,
                                  _Float16* __restrict__ dst, int n) {
  int i = blockIdx.x * 256 + threadIdx.x;
  if (i < n) dst[i] = (_Float16)src[i];
}

// Per-row L2 norm scaling: xn = x / max(||x||/sqrt(D), eps) * gamma.
// Also emits x itself as f16 for the KV projection.
__global__ __launch_bounds__(256) void norm_kernel(const float* __restrict__ x,
                                                   const float* __restrict__ gamma,
                                                   _Float16* __restrict__ xn_h,
                                                   _Float16* __restrict__ x_h) {
  __shared__ float red[8];
  int row = blockIdx.x;
  int t = threadIdx.x;
  const float* xr = x + (size_t)row * DQ;
  float v[4];
  float ss = 0.f;
#pragma unroll
  for (int i = 0; i < 4; ++i) {
    v[i] = xr[i * 256 + t];
    ss += v[i] * v[i];
  }
#pragma unroll
  for (int m = 16; m >= 1; m >>= 1) ss += __shfl_xor(ss, m, 32);
  if ((t & 31) == 0) red[t >> 5] = ss;
  __syncthreads();
  float tot = 0.f;
#pragma unroll
  for (int i = 0; i < 8; ++i) tot += red[i];
  float norm = sqrtf(tot) * 0.03125f;          // * D^-0.5
  float inv = 1.0f / fmaxf(norm, 1e-8f);
#pragma unroll
  for (int i = 0; i < 4; ++i) {
    int c = i * 256 + t;
    xn_h[(size_t)row * DQ + c] = (_Float16)(v[i] * inv * gamma[c]);
    x_h[(size_t)row * DQ + c] = (_Float16)v[i];
  }
}

// RoPE + head split: src [B,N,row_stride] f16 -> dst [B,H,N,DH] f16.
__global__ __launch_bounds__(256) void rope_split_kernel(
    const _Float16* __restrict__ src, int row_stride, int col_off,
    _Float16* __restrict__ dst, const float* __restrict__ pos, float scale,
    int do_rope) {
  int tid = blockIdx.x * 256 + threadIdx.x;
  int d = tid & 63;
  int n = (tid >> 6) & (NQ - 1);
  int h = (tid >> 17) & (HQ - 1);
  int b = tid >> 21;
  const _Float16* srow =
      src + (size_t)(b * NQ + n) * row_stride + col_off + h * DHQ;
  float t = (float)srow[d];
  float out;
  if (do_rope) {
    float f = pos[n * DHQ + d];
    float partner = (float)srow[d ^ 32];
    float rot = (d < 32) ? -partner : partner;  // rotate_half
    out = t * __cosf(f) + rot * __sinf(f);
  } else {
    out = t;
  }
  dst[(((size_t)(b * HQ + h)) * NQ + n) * DHQ + d] = (_Float16)(out * scale);
}

// ---------------- WMMA GEMM: C[M,N] = A[M,K] * B[K,N] (+bias) ----------------
// Block tile 128x128, 8 waves in 4x2, each wave 32x64 (2x4 WMMA tiles),
// K step 32. A slab: async global->LDS. B slab: transposed staging so all
// fragment reads are contiguous (vectorized ds_load_b128).

template <typename OutT>
__global__ __launch_bounds__(256) void gemm_f16_kernel(
    const _Float16* __restrict__ A, const _Float16* __restrict__ Bw,
    OutT* __restrict__ C, const float* __restrict__ bias, int M, int Nn,
    int Kk) {
  __shared__ _Float16 Al[128][40];
  __shared__ _Float16 Bt[128][40];  // Bt[n][k] = B[k][n] for current slab
  int t = threadIdx.x;
  int w = t >> 5, lane = t & 31;
  int wm = (w >> 1) * 32, wn = (w & 1) * 64;
  int m0 = blockIdx.y * 128, n0 = blockIdx.x * 128;
  v8f acc[2][4] = {};

  for (int k0 = 0; k0 < Kk; k0 += 32) {
    __syncthreads();
    {  // A slab 128x32: 16 halfs per thread via async global->LDS
      int r = t >> 1, s = (t & 1) * 16;
      const _Float16* g = A + (size_t)(m0 + r) * Kk + k0 + s;
      async_copy_b128(lds_addr32(&Al[r][s]), g);
      async_copy_b128(lds_addr32(&Al[r][s + 8]), g + 8);
    }
    {  // B slab 32x128 -> transposed into Bt: 16 halfs per thread
      int r = t >> 3, c = (t & 7) * 16;
      const _Float16* g = Bw + (size_t)(k0 + r) * Nn + n0 + c;
      h8 b0 = *(const h8*)g;
      h8 b1 = *(const h8*)(g + 8);
#pragma unroll
      for (int i = 0; i < 8; ++i) Bt[c + i][r] = b0[i];
#pragma unroll
      for (int i = 0; i < 8; ++i) Bt[c + 8 + i][r] = b1[i];
    }
    if (k0 + 32 < Kk) {  // prefetch next B slab (global_prefetch_b8)
      __builtin_prefetch(Bw + (size_t)(k0 + 32 + (t >> 3)) * Nn + n0, 0, 0);
    }
    async_wait0();
    __syncthreads();
    v16h a0 = load_frag_a(&Al[wm][0], 40);
    v16h a1 = load_frag_a(&Al[wm + 16][0], 40);
#pragma unroll
    for (int j = 0; j < 4; ++j) {
      v16h b = load_frag_bT(&Bt[wn + j * 16][0], 40);
      acc[0][j] = wmma32(a0, b, acc[0][j]);
      acc[1][j] = wmma32(a1, b, acc[1][j]);
    }
  }

  int n = lane & 15, mb = (lane >> 4) * 8;
#pragma unroll
  for (int i = 0; i < 2; ++i)
#pragma unroll
    for (int j = 0; j < 4; ++j)
#pragma unroll
      for (int r = 0; r < 8; ++r) {
        int row = m0 + wm + i * 16 + mb + r;
        int col = n0 + wn + j * 16 + n;
        float v = acc[i][j][r];
        if (bias) v += bias[col];
        C[(size_t)row * Nn + col] = (OutT)v;
      }
}

// ---------------- Flash attention (causal, online softmax) ----------------
// Grid: (N/128, B*H). 8 waves x 16 query rows. Keys streamed 64 at a time.
// K block: async global->LDS (row-major feeds Q*K^T directly).
// V block: staged transposed so P*V fragment reads are contiguous.

__global__ __launch_bounds__(256) void flash_attn_kernel(
    const _Float16* __restrict__ Q, const _Float16* __restrict__ Kh,
    const _Float16* __restrict__ Vh, _Float16* __restrict__ Om) {
  __shared__ _Float16 Kl[64][72];      // Kl[key][dh]
  __shared__ _Float16 Vt[64][72];      // Vt[dh][key] (transposed)
  __shared__ _Float16 Pl[8][16][72];   // per-wave softmax tile

  int bh = blockIdx.y;
  int b = bh >> 4, h = bh & 15;
  int rb = blockIdx.x;
  int t = threadIdx.x;
  int w = t >> 5, lane = t & 31;
  int n = lane & 15, mb = (lane >> 4) * 8;

  size_t headoff = (size_t)(b * HQ + h) * NQ * DHQ;
  int qm0 = rb * 128 + w * 16;

  // Q fragments for this wave's 16 rows (two K-chunks of 32 over DH=64)
  v16h aq0 = load_frag_a(Q + headoff + (size_t)qm0 * DHQ + 0, DHQ);
  v16h aq1 = load_frag_a(Q + headoff + (size_t)qm0 * DHQ + 32, DHQ);

  v8f acc[4] = {};
  float mrow[8], lrow[8];
#pragma unroll
  for (int r = 0; r < 8; ++r) { mrow[r] = -1e30f; lrow[r] = 0.f; }

  int nkb = (rb + 1) * 2;  // causal: keys up to end of this 128-row tile
  for (int kb = 0; kb < nkb; ++kb) {
    __syncthreads();
    {  // K block 64x64: async copy; V block: transposed register staging
      int r = t >> 2, s = (t & 3) * 16;
      const _Float16* gk = Kh + headoff + (size_t)(kb * 64 + r) * DHQ + s;
      async_copy_b128(lds_addr32(&Kl[r][s]), gk);
      async_copy_b128(lds_addr32(&Kl[r][s + 8]), gk + 8);
      const _Float16* gv = Vh + headoff + (size_t)(kb * 64 + r) * DHQ + s;
      h8 v0 = *(const h8*)gv;
      h8 v1 = *(const h8*)(gv + 8);
#pragma unroll
      for (int i = 0; i < 8; ++i) Vt[s + i][r] = v0[i];
#pragma unroll
      for (int i = 0; i < 8; ++i) Vt[s + 8 + i][r] = v1[i];
    }
    async_wait0();
    __syncthreads();

    // S = Q * K^T : 16 x 64 scores in 4 tiles (B(k=dh, n=key) = Kl[key][dh])
    v8f s4[4];
#pragma unroll
    for (int nt = 0; nt < 4; ++nt) {
      v8f z = {};
      z = wmma32(aq0, load_frag_bT(&Kl[nt * 16][0], 72), z);
      z = wmma32(aq1, load_frag_bT(&Kl[nt * 16][32], 72), z);
      s4[nt] = z;
    }

    // causal mask + online softmax stats (row spans lanes 0..15 of half-wave)
    float alpha[8];
#pragma unroll
    for (int r = 0; r < 8; ++r) {
      int row_g = qm0 + mb + r;
      float lm = -1e30f;
#pragma unroll
      for (int nt = 0; nt < 4; ++nt) {
        int key = kb * 64 + nt * 16 + n;
        if (key > row_g) s4[nt][r] = -1e30f;
        lm = fmaxf(lm, s4[nt][r]);
      }
      lm = fmaxf(lm, __shfl_xor(lm, 1, 32));
      lm = fmaxf(lm, __shfl_xor(lm, 2, 32));
      lm = fmaxf(lm, __shfl_xor(lm, 4, 32));
      lm = fmaxf(lm, __shfl_xor(lm, 8, 32));
      float mn = fmaxf(mrow[r], lm);
      float al = __expf(mrow[r] - mn);
      float ps = 0.f;
#pragma unroll
      for (int nt = 0; nt < 4; ++nt) {
        float p = __expf(s4[nt][r] - mn);
        Pl[w][mb + r][nt * 16 + n] = (_Float16)p;
        ps += p;
      }
      ps += __shfl_xor(ps, 1, 32);
      ps += __shfl_xor(ps, 2, 32);
      ps += __shfl_xor(ps, 4, 32);
      ps += __shfl_xor(ps, 8, 32);
      lrow[r] = lrow[r] * al + ps;
      mrow[r] = mn;
      alpha[r] = al;
    }
#pragma unroll
    for (int dt = 0; dt < 4; ++dt)
#pragma unroll
      for (int r = 0; r < 8; ++r) acc[dt][r] *= alpha[r];

    // O += P * V  with B(k=key, n=dh) = Vt[dh][key] (contiguous loads)
    v16h ap0 = load_frag_a(&Pl[w][0][0], 72);
    v16h ap1 = load_frag_a(&Pl[w][0][32], 72);
#pragma unroll
    for (int dt = 0; dt < 4; ++dt) {
      acc[dt] = wmma32(ap0, load_frag_bT(&Vt[dt * 16][0], 72), acc[dt]);
      acc[dt] = wmma32(ap1, load_frag_bT(&Vt[dt * 16][32], 72), acc[dt]);
    }
  }

  // epilogue: O /= l, store merged [B, N, H*DH] as f16 for out-projection
#pragma unroll
  for (int r = 0; r < 8; ++r) {
    float inv = 1.0f / lrow[r];
    int row_g = qm0 + mb + r;
#pragma unroll
    for (int dt = 0; dt < 4; ++dt) {
      Om[(size_t)(b * NQ + row_g) * (HQ * DHQ) + h * DHQ + dt * 16 + n] =
          (_Float16)(acc[dt][r] * inv);
    }
  }
}

// ---------------- host orchestration ----------------

extern "C" void kernel_launch(void* const* d_in, const int* in_sizes, int n_in,
                              void* d_out, int out_size, void* d_ws,
                              size_t ws_size, hipStream_t stream) {
  (void)in_sizes; (void)n_in; (void)out_size; (void)ws_size;
  const float* x     = (const float*)d_in[0];  // [2,2048,1024]
  const float* pos   = (const float*)d_in[1];  // [1,1,2048,64]
  const float* gamma = (const float*)d_in[2];  // [1024]
  const float* Wq    = (const float*)d_in[3];  // [1024,1024]
  const float* Wkv   = (const float*)d_in[4];  // [1024,2048]
  const float* Wo    = (const float*)d_in[5];  // [1024,1024]
  const float* bo    = (const float*)d_in[6];  // [1024]
  float* out = (float*)d_out;

  char* ws = (char*)d_ws;
  const size_t MB = 1u << 20;
  _Float16* Wq_h   = (_Float16*)(ws + 0 * MB);   // 2 MB
  _Float16* Wkv_h  = (_Float16*)(ws + 2 * MB);   // 4 MB
  _Float16* Wo_h   = (_Float16*)(ws + 6 * MB);   // 2 MB
  _Float16* x_h    = (_Float16*)(ws + 8 * MB);   // 8 MB
  _Float16* xn_h   = (_Float16*)(ws + 16 * MB);  // 8 MB
  _Float16* q_lin  = (_Float16*)(ws + 24 * MB);  // 8 MB
  _Float16* kv_lin = (_Float16*)(ws + 32 * MB);  // 16 MB
  _Float16* q_hd   = (_Float16*)(ws + 48 * MB);  // 8 MB
  _Float16* k_hd   = (_Float16*)(ws + 56 * MB);  // 8 MB
  _Float16* v_hd   = (_Float16*)(ws + 64 * MB);  // 8 MB
  _Float16* attn_m = (_Float16*)(ws + 72 * MB);  // 8 MB

  // weight conversion to f16
  f32_to_f16_kernel<<<(DQ * DQ) / 256, 256, 0, stream>>>(Wq, Wq_h, DQ * DQ);
  f32_to_f16_kernel<<<(DQ * 2 * DQ) / 256, 256, 0, stream>>>(Wkv, Wkv_h,
                                                             DQ * 2 * DQ);
  f32_to_f16_kernel<<<(DQ * DQ) / 256, 256, 0, stream>>>(Wo, Wo_h, DQ * DQ);

  // norm + x cast
  norm_kernel<<<MROWS, 256, 0, stream>>>(x, gamma, xn_h, x_h);

  // projections: q = xn @ Wq ; kv = x @ Wkv   (block tile 128x128)
  gemm_f16_kernel<_Float16><<<dim3(DQ / 128, MROWS / 128), 256, 0, stream>>>(
      xn_h, Wq_h, q_lin, nullptr, MROWS, DQ, DQ);
  gemm_f16_kernel<_Float16><<<dim3(2 * DQ / 128, MROWS / 128), 256, 0, stream>>>(
      x_h, Wkv_h, kv_lin, nullptr, MROWS, 2 * DQ, DQ);

  // RoPE + head split (q also scaled by DH^-0.5)
  const int rope_blocks = (BQ * HQ * NQ * DHQ) / 256;  // 16384
  rope_split_kernel<<<rope_blocks, 256, 0, stream>>>(q_lin, DQ, 0, q_hd, pos,
                                                     0.125f, 1);
  rope_split_kernel<<<rope_blocks, 256, 0, stream>>>(kv_lin, 2 * DQ, 0, k_hd,
                                                     pos, 1.0f, 1);
  rope_split_kernel<<<rope_blocks, 256, 0, stream>>>(kv_lin, 2 * DQ, DQ, v_hd,
                                                     pos, 1.0f, 0);

  // flash attention
  flash_attn_kernel<<<dim3(NQ / 128, BQ * HQ), 256, 0, stream>>>(q_hd, k_hd,
                                                                 v_hd, attn_m);

  // output projection + bias -> f32
  gemm_f16_kernel<float><<<dim3(DQ / 128, MROWS / 128), 256, 0, stream>>>(
      attn_m, Wo_h, out, bo, MROWS, DQ, DQ);
}
